// GcnDenseModel_43095701848635
// MI455X (gfx1250) — matrix operand order repaired and checked by reading, not compiled, not run on hardware
//
#include <hip/hip_runtime.h>

// ---------------------------------------------------------------------------
// GCN + Transformer encoder for MI455X (gfx1250), bf16 WMMA everywhere.
// B=64 N=512 E=2048 FIN=300 D=256 A=64 NH=4 dh=64
// GEMMs: double-buffered LDS staging via global_load_async_to_lds_b128.
// ---------------------------------------------------------------------------

typedef __attribute__((ext_vector_type(16))) __bf16 v16bf;
typedef __attribute__((ext_vector_type(8)))  float  v8f;
typedef __attribute__((ext_vector_type(4)))  unsigned int u32x4;

#define CEILDIV(a,b) (((a)+(b)-1)/(b))

static constexpr int Bc   = 64;
static constexpr int Nc   = 512;
static constexpr int Ec   = 2048;
static constexpr int FINc = 300;
static constexpr int Dc   = 256;
static constexpr int NHc  = 4;
static constexpr int BNr  = Bc * Nc;   // 32768 node rows
static constexpr int BEr  = Bc * Ec;   // 131072 edges

__device__ __forceinline__ unsigned short f2bf(float f) {
    unsigned int u = __float_as_uint(f);
    u += 0x7FFFu + ((u >> 16) & 1u);            // round-to-nearest-even
    return (unsigned short)(u >> 16);
}

struct B32 { u32x4 a, b; };

__device__ __forceinline__ v16bf ld2(const unsigned short* p0, const unsigned short* p1) {
    B32 u;
    u.a = *(const u32x4*)p0;
    u.b = *(const u32x4*)p1;
    return __builtin_bit_cast(v16bf, u);
}

__device__ __forceinline__ v8f wmma_bf16(v16bf a, v16bf b, v8f c) {
    return __builtin_amdgcn_wmma_f32_16x16x32_bf16(false, a, false, b, (short)0, c, false, false);
}

// async DMA: 16B per lane, global (saddr + 32-bit vaddr) -> LDS[ldsaddr]
__device__ __forceinline__ void async_ld16(unsigned lds, unsigned gofs, const void* base) {
    asm volatile("global_load_async_to_lds_b128 %0, %1, %2"
                 :: "v"(lds), "v"(gofs), "s"(base) : "memory");
}
__device__ __forceinline__ void async_ld16_off16(unsigned lds, unsigned gofs, const void* base) {
    asm volatile("global_load_async_to_lds_b128 %0, %1, %2 offset:16"
                 :: "v"(lds), "v"(gofs), "s"(base) : "memory");
}
__device__ __forceinline__ void wait_async0() {
    asm volatile("s_wait_asynccnt 0" ::: "memory");
}
__device__ __forceinline__ unsigned lds_off(const void* p) {
    return (unsigned)(unsigned long long)p;   // shared aperture: addr[31:0] == LDS offset
}

// ---------------------------------------------------------------------------
// Elementwise / prep kernels
// ---------------------------------------------------------------------------
__global__ void k_zero(float* p, int n) {
    int i = blockIdx.x * 256 + threadIdx.x;
    if (i < n) p[i] = 0.f;
}
__global__ void k_fill(float* p, float v, int n) {
    int i = blockIdx.x * 256 + threadIdx.x;
    if (i < n) p[i] = v;
}
__global__ void k_cast(const float* __restrict__ s, unsigned short* __restrict__ d, int n) {
    int i = blockIdx.x * 256 + threadIdx.x;
    if (i < n) d[i] = f2bf(s[i]);
}
// x (BN,300) -> bf16 padded to (BN,320)
__global__ void k_pad_x(const float* __restrict__ x, unsigned short* __restrict__ xb) {
    int i = blockIdx.x * 256 + threadIdx.x;
    if (i >= BNr * 320) return;
    int r = i / 320, c = i - r * 320;
    xb[i] = (c < FINc) ? f2bf(x[(size_t)r * FINc + c]) : (unsigned short)0;
}
// W13[d,f] = sum_k W3[d, k] * W1[k, f]   (W3a . W1), padded to 320 cols
__global__ void k_combine13(const float* __restrict__ W3, const float* __restrict__ W1,
                            unsigned short* __restrict__ w13) {
    int i = blockIdx.x * 256 + threadIdx.x;
    if (i >= 256 * 320) return;
    int d = i / 320, f = i - d * 320;
    if (f >= FINc) { w13[i] = 0; return; }
    float s = 0.f;
    for (int k = 0; k < 256; ++k) s += W3[(size_t)d * 512 + k] * W1[(size_t)k * FINc + f];
    w13[i] = f2bf(s);
}
// W23[d,a] = sum_k W3[d, 256+k] * W2[k, a]   (W3b . W2)
__global__ void k_combine23(const float* __restrict__ W3, const float* __restrict__ W2,
                            unsigned short* __restrict__ w23) {
    int i = blockIdx.x * 256 + threadIdx.x;
    if (i >= 256 * 64) return;
    int d = i >> 6, a = i & 63;
    float s = 0.f;
    for (int k = 0; k < 256; ++k) s += W3[(size_t)d * 512 + 256 + k] * W2[(size_t)k * 64 + a];
    w23[i] = f2bf(s);
}
// selfbias[d] = b3[d] + sum_k W3[d, 256+k]  (self-loop edge_attr == ones)
__global__ void k_selfbias(const float* __restrict__ W3, const float* __restrict__ b3,
                           float* __restrict__ sb) {
    int d = threadIdx.x;
    float s = b3[d];
    for (int k = 0; k < 256; ++k) s += W3[(size_t)d * 512 + 256 + k];
    sb[d] = s;
}
// flatten edge indices to global node ids + degree atomics (deg pre-filled 1.0 for self loop)
__global__ void k_edge_prep(const int* __restrict__ er_in, const int* __restrict__ ec_in,
                            int* __restrict__ er, int* __restrict__ ec, float* __restrict__ deg) {
    int i = blockIdx.x * 256 + threadIdx.x;
    if (i >= BEr) return;
    int b = i >> 11;                       // E = 2048
    int r = er_in[i] + (b << 9);
    int c = ec_in[i] + (b << 9);
    er[i] = r; ec[i] = c;
    atomicAdd(&deg[c], 1.0f);
}
__global__ void k_dinv(const float* __restrict__ deg, float* __restrict__ dinv, int n) {
    int i = blockIdx.x * 256 + threadIdx.x;
    if (i < n) dinv[i] = rsqrtf(deg[i]);
}
// h = tanh((agg_edges + selfmsg)/deg + gcn_bias) * mask
__global__ void k_gcn_final(const float* __restrict__ agg, const float* __restrict__ Y1,
                            const float* __restrict__ sb, const float* __restrict__ deg,
                            const float* __restrict__ dinv, const float* __restrict__ gcn_bias,
                            const int* __restrict__ lengths, float* __restrict__ h) {
    int i = blockIdx.x * 256 + threadIdx.x;
    if (i >= BNr * 256) return;
    int r = i >> 8, c = i & 255;
    int b = r >> 9, node = r & 511;
    float di = dinv[r];
    float val = (agg[i] + (Y1[i] + sb[c]) * di * di) / deg[r] + gcn_bias[c];
    val = tanhf(val);
    if (node >= lengths[b]) val = 0.f;
    h[i] = val;
}
// per-channel batchnorm stats over 32768 rows (coalesced; atomics once per block)
__global__ __launch_bounds__(256) void k_bn_stats(const float* __restrict__ in,
                                                  float* __restrict__ sums, float* __restrict__ sumsq) {
    int c = threadIdx.x;
    int r0 = blockIdx.x * 128;
    float s = 0.f, s2 = 0.f;
    for (int i = 0; i < 128; ++i) {
        float v = in[(size_t)(r0 + i) * 256 + c];
        s += v; s2 += v * v;
    }
    atomicAdd(&sums[c], s);
    atomicAdd(&sumsq[c], s2);
}
__global__ void k_bn_fin(const float* __restrict__ sums, const float* __restrict__ sumsq,
                         const float* __restrict__ g, const float* __restrict__ bb,
                         float* __restrict__ scale, float* __restrict__ shift, float invN) {
    int c = threadIdx.x;
    float m = sums[c] * invN;
    float v = sumsq[c] * invN - m * m;
    float sc = g[c] * rsqrtf(v + 1e-5f);
    scale[c] = sc;
    shift[c] = bb[c] - m * sc;
}
__global__ void k_bn_apply(const float* __restrict__ h, const float* __restrict__ scale,
                           const float* __restrict__ shift, float* __restrict__ outf,
                           unsigned short* __restrict__ outh) {
    int i = blockIdx.x * 256 + threadIdx.x;
    if (i >= BNr * 256) return;
    int c = i & 255;
    float v = h[i] * scale[c] + shift[c];
    outf[i] = v;
    outh[i] = f2bf(v);
}
// row LayerNorm over D=256 (in already contains residual sum)
__global__ __launch_bounds__(256) void k_layernorm(const float* __restrict__ in,
                                                   const float* __restrict__ g, const float* __restrict__ bb,
                                                   float* __restrict__ outf, unsigned short* outh) {
    __shared__ float red[256];
    int r = blockIdx.x, t = threadIdx.x;
    float v = in[(size_t)r * 256 + t];
    red[t] = v; __syncthreads();
    for (int s = 128; s > 0; s >>= 1) { if (t < s) red[t] += red[t + s]; __syncthreads(); }
    float mean = red[0] * (1.f / 256.f);
    __syncthreads();
    float d = v - mean;
    red[t] = d * d; __syncthreads();
    for (int s = 128; s > 0; s >>= 1) { if (t < s) red[t] += red[t + s]; __syncthreads(); }
    float var = red[0] * (1.f / 256.f);
    float o = d * rsqrtf(var + 1e-5f) * g[t] + bb[t];
    outf[(size_t)r * 256 + t] = o;
    if (outh) outh[(size_t)r * 256 + t] = f2bf(o);
}
// masked mean/max pooling with fused bn2 affine
__global__ __launch_bounds__(256) void k_pool(const float* __restrict__ h2,
                                              const float* __restrict__ scale, const float* __restrict__ shift,
                                              const int* __restrict__ lengths, float* __restrict__ pool) {
    int b = blockIdx.x, c = threadIdx.x;
    int len = lengths[b];
    float sc = scale[c], sh = shift[c];
    float s = 0.f, mx = -3.4e38f;
    for (int n = 0; n < len; ++n) {
        float v = h2[(size_t)(b * 512 + n) * 256 + c] * sc + sh;
        s += v; mx = fmaxf(mx, v);
    }
    pool[b * 512 + c]       = mx;
    pool[b * 512 + 256 + c] = s / (float)len;
}
// final linear (64x2, K=512) + batchnorm over batch + softmax, single block
__global__ __launch_bounds__(128) void k_final(const float* __restrict__ pool,
                                               const float* __restrict__ lw, const float* __restrict__ lb,
                                               const float* __restrict__ g3, const float* __restrict__ b3n,
                                               float* __restrict__ out) {
    __shared__ float resb[128];
    __shared__ float ss[4];
    __shared__ float zb[128];
    int t = threadIdx.x, b = t >> 1, cls = t & 1;
    float s = 0.f;
    for (int k = 0; k < 512; ++k) s += pool[b * 512 + k] * lw[cls * 512 + k];
    s += lb[cls];
    resb[t] = s;
    __syncthreads();
    if (t < 2) {
        float m = 0.f, v = 0.f;
        for (int i = 0; i < 64; ++i) m += resb[i * 2 + t];
        m *= (1.f / 64.f);
        for (int i = 0; i < 64; ++i) { float x = resb[i * 2 + t] - m; v += x * x; }
        v *= (1.f / 64.f);
        float sc = g3[t] * rsqrtf(v + 1e-5f);
        ss[t] = sc; ss[2 + t] = b3n[t] - m * sc;
    }
    __syncthreads();
    zb[t] = resb[t] * ss[cls] + ss[2 + cls];
    __syncthreads();
    float z0 = zb[b * 2], z1 = zb[b * 2 + 1];
    float mx = fmaxf(z0, z1);
    float e0 = __expf(z0 - mx), e1 = __expf(z1 - mx);
    out[t] = ((cls == 0) ? e0 : e1) / (e0 + e1);
}

// ---------------------------------------------------------------------------
// bf16 WMMA GEMM: C[M,N] = A[M,K] (row-major) * W[N,K]^T (row-major) + epi
// block 256 threads (8 waves), tile 64x128, K step 32; wave -> 16x64 strip.
// LDS staging is double-buffered via global_load_async_to_lds_b128 (ASYNCcnt):
// tile i+1 streams into LDS while tile i feeds the WMMAs.
// EPI: 0 outf=acc+bias | 1 outf=acc+bias+resid | 2 outh=bf16(relu(acc+bias))
//      3 edge scatter into agg | 4 qkv layout split (scale q by 0.125)
// ---------------------------------------------------------------------------
template <int EPI>
__global__ __launch_bounds__(256) void gemm_bf16(
        const unsigned short* __restrict__ A, int lda,
        const unsigned short* __restrict__ W, int ldw,
        const float* __restrict__ bias,
        float* __restrict__ outf, unsigned short* __restrict__ outh, int ldo,
        const float* __restrict__ resid,
        const int* __restrict__ erow, const int* __restrict__ ecol,
        const float* __restrict__ dinv, float* __restrict__ agg,
        const float* __restrict__ Y1, const float* __restrict__ b3,
        unsigned short* __restrict__ qb, unsigned short* __restrict__ kb,
        unsigned short* __restrict__ vtb,
        int K)
{
    __shared__ alignas(16) unsigned short As[2][64 * 32];
    __shared__ alignas(16) unsigned short Ws[2][128 * 32];

    const int tid  = threadIdx.x;
    const int rowBase = blockIdx.y * 64;
    const int colBase = blockIdx.x * 128;
    const int wave = tid >> 5, lane = tid & 31;
    const int wr = wave & 3, wc = wave >> 2;
    const int lm = lane & 15;
    const int lhA = (lane < 16) ? 0 : 8;     // A operand: K 0-7/16-23 vs 8-15/24-31
    const int lhB = (lane < 16) ? 0 : 16;    // unused; B is loaded with lhA (same K order as A)

    // per-thread DMA addressing (16B A chunk, 32B W chunk per thread per tile)
    const unsigned ldsA = lds_off(&As[0][(tid >> 2) * 32 + (tid & 3) * 8]);
    const unsigned ldsW = lds_off(&Ws[0][(tid >> 1) * 32 + (tid & 1) * 16]);
    const unsigned strA = (unsigned)sizeof(As[0]);
    const unsigned strW = (unsigned)sizeof(Ws[0]);
    unsigned gA = (unsigned)(((size_t)(rowBase + (tid >> 2)) * lda + (tid & 3) * 8) * 2);
    unsigned gW = (unsigned)(((size_t)(colBase + (tid >> 1)) * ldw + (tid & 1) * 16) * 2);

    v8f acc[4];
    #pragma unroll
    for (int i = 0; i < 4; ++i) acc[i] = {};

    // prologue: stream tile 0 into buffer 0
    async_ld16(ldsA, gA, A);
    async_ld16(ldsW, gW, W);
    async_ld16_off16(ldsW, gW, W);

    int buf = 0;
    for (int k0 = 0; k0 < K; k0 += 32) {
        wait_async0();          // this wave's portion of the current tile landed
        __syncthreads();        // every wave's portion landed
        if (k0 + 32 < K) {      // overlap: stream next tile into alternate buffer
            gA += 64; gW += 64;
            int nb = buf ^ 1;
            async_ld16(ldsA + nb * strA, gA, A);
            async_ld16(ldsW + nb * strW, gW, W);
            async_ld16_off16(ldsW + nb * strW, gW, W);
        }
        const unsigned short* Ab = &As[buf][0];
        const unsigned short* Wb = &Ws[buf][0];
        int ar = wr * 16 + lm;
        v16bf av = ld2(&Ab[ar * 32 + lhA], &Ab[ar * 32 + 16 + lhA]);
        #pragma unroll
        for (int nt = 0; nt < 4; ++nt) {
            int wrow = wc * 64 + nt * 16 + lm;
            v16bf bv = ld2(&Wb[wrow * 32 + lhA], &Wb[wrow * 32 + 16 + lhA]);
            acc[nt] = wmma_bf16(av, bv, acc[nt]);
        }
        buf ^= 1;
    }

    // epilogue: lane holds column n=lm, rows mofs..mofs+7 of its 16x16 tile
    const int mofs = (lane < 16) ? 0 : 8;
    #pragma unroll
    for (int nt = 0; nt < 4; ++nt) {
        int c = colBase + wc * 64 + nt * 16 + lm;
        float bval = (EPI == 3) ? 0.f : (bias ? bias[c] : 0.f);
        #pragma unroll
        for (int j = 0; j < 8; ++j) {
            int r = rowBase + wr * 16 + mofs + j;
            float v = acc[nt][j] + bval;
            if (EPI == 0) {
                outf[(size_t)r * ldo + c] = v;
            } else if (EPI == 1) {
                outf[(size_t)r * ldo + c] = v + resid[(size_t)r * ldo + c];
            } else if (EPI == 2) {
                outh[(size_t)r * ldo + c] = f2bf(fmaxf(v, 0.f));
            } else if (EPI == 3) {
                int er = erow[r], ec = ecol[r];
                float nrm = dinv[er] * dinv[ec];
                float msg = (acc[nt][j] + Y1[(size_t)er * 256 + c] + b3[c]) * nrm;
                atomicAdd(&agg[(size_t)ec * 256 + c], msg);
            } else { // EPI == 4: qkv split, attention-friendly layouts
                int b = r >> 9, node = r & 511;
                if (c < 256) {
                    int hh = c >> 6, d = c & 63;
                    qb[((size_t)(b * NHc + hh) * 512 + node) * 64 + d] = f2bf(v * 0.125f);
                } else if (c < 512) {
                    int c2 = c - 256, hh = c2 >> 6, d = c2 & 63;
                    kb[((size_t)(b * NHc + hh) * 512 + node) * 64 + d] = f2bf(v);
                } else {
                    int c2 = c - 512, hh = c2 >> 6, d = c2 & 63;
                    vtb[((size_t)(b * NHc + hh) * 64 + d) * 512 + node] = f2bf(v);
                }
            }
        }
    }
}

// ---------------------------------------------------------------------------
// Attention: one wave per (b, head, 16-query tile). Full 512-key softmax
// (reference does not mask attention). grid = B*NH*32, block = 32.
// ---------------------------------------------------------------------------
__global__ __launch_bounds__(32) void k_attention(const unsigned short* __restrict__ qb,
                                                  const unsigned short* __restrict__ kb,
                                                  const unsigned short* __restrict__ vtb,
                                                  unsigned short* __restrict__ attnb)
{
    __shared__ alignas(16) float sc[16][512];
    __shared__ alignas(16) unsigned short pb[16][512];

    const int lane = threadIdx.x;
    const int idx = blockIdx.x;
    const int qt = idx & 31;
    const int bh = idx >> 5;
    const int b = bh >> 2, hh = bh & 3;
    const int lm = lane & 15;
    const int lhA = (lane < 16) ? 0 : 8;
    const int lhB = (lane < 16) ? 0 : 16;
    const int mofs = (lane < 16) ? 0 : 8;

    // q tile (16x64) -> A operands for k0=0 and k0=32 (scale folded in at qkv)
    const unsigned short* qrow = qb + (size_t)(bh * 512 + qt * 16 + lm) * 64;
    v16bf a0 = ld2(qrow + lhA,      qrow + 16 + lhA);
    v16bf a1 = ld2(qrow + 32 + lhA, qrow + 48 + lhA);

    // scores: S(16x512) = q . K^T
    for (int kt = 0; kt < 32; ++kt) {
        const unsigned short* krow = kb + (size_t)(bh * 512 + kt * 16 + lm) * 64;
        v16bf b0 = ld2(krow + lhA,      krow + 16 + lhA);
        v16bf b1 = ld2(krow + 32 + lhA, krow + 48 + lhA);
        v8f acc = {};
        acc = wmma_bf16(a0, b0, acc);
        acc = wmma_bf16(a1, b1, acc);
        #pragma unroll
        for (int j = 0; j < 8; ++j) sc[mofs + j][kt * 16 + lm] = acc[j];
    }
    __syncthreads();

    // softmax each of 16 rows across 512 keys (lane-parallel + shuffle reduce)
    for (int m = 0; m < 16; ++m) {
        float mx = -3.4e38f;
        for (int t = lane; t < 512; t += 32) mx = fmaxf(mx, sc[m][t]);
        for (int off = 16; off > 0; off >>= 1) mx = fmaxf(mx, __shfl_xor(mx, off, 32));
        float s = 0.f;
        for (int t = lane; t < 512; t += 32) {
            float e = __expf(sc[m][t] - mx);
            sc[m][t] = e; s += e;
        }
        for (int off = 16; off > 0; off >>= 1) s += __shfl_xor(s, off, 32);
        float inv = 1.f / s;
        for (int t = lane; t < 512; t += 32) pb[m][t] = f2bf(sc[m][t] * inv);
    }
    __syncthreads();

    // O(16x64) = P(16x512) . V(512x64), vT stored d-major for contiguous B loads
    for (int dt = 0; dt < 4; ++dt) {
        v8f acc = {};
        for (int ks = 0; ks < 16; ++ks) {
            int k0 = ks * 32;
            v16bf av = ld2(&pb[lm][k0 + lhA], &pb[lm][k0 + 16 + lhA]);
            const unsigned short* vrow = vtb + (size_t)(bh * 64 + dt * 16 + lm) * 512 + k0;
            v16bf bv = ld2(vrow + lhA, vrow + 16 + lhA);
            acc = wmma_bf16(av, bv, acc);
        }
        #pragma unroll
        for (int j = 0; j < 8; ++j) {
            int m = mofs + j;
            attnb[(size_t)(b * 512 + qt * 16 + m) * 256 + hh * 64 + dt * 16 + lm] = f2bf(acc[j]);
        }
    }
}

// ---------------------------------------------------------------------------
// Host orchestration
// ---------------------------------------------------------------------------
extern "C" void kernel_launch(void* const* d_in, const int* in_sizes, int n_in,
                              void* d_out, int out_size, void* d_ws, size_t ws_size,
                              hipStream_t stream) {
    const float* x          = (const float*)d_in[0];
    const float* edge_attr  = (const float*)d_in[1];
    const int*   edge_row   = (const int*)d_in[2];
    const int*   edge_col   = (const int*)d_in[3];
    const int*   lengths    = (const int*)d_in[4];
    const float* W1         = (const float*)d_in[5];
    const float* W2         = (const float*)d_in[6];
    const float* W3         = (const float*)d_in[7];
    const float* b3         = (const float*)d_in[8];
    const float* gcn_bias   = (const float*)d_in[9];
    const float* bn1_g      = (const float*)d_in[10];
    const float* bn1_b      = (const float*)d_in[11];
    const float* in_proj_w  = (const float*)d_in[12];
    const float* in_proj_b  = (const float*)d_in[13];
    const float* out_proj_w = (const float*)d_in[14];
    const float* out_proj_b = (const float*)d_in[15];
    const float* ln1_g      = (const float*)d_in[16];
    const float* ln1_b      = (const float*)d_in[17];
    const float* lin1_w     = (const float*)d_in[18];
    const float* lin1_b     = (const float*)d_in[19];
    const float* lin2_w     = (const float*)d_in[20];
    const float* lin2_b     = (const float*)d_in[21];
    const float* ln2_g      = (const float*)d_in[22];
    const float* ln2_b      = (const float*)d_in[23];
    const float* bn2_g      = (const float*)d_in[24];
    const float* bn2_b      = (const float*)d_in[25];
    const float* lin_w      = (const float*)d_in[26];
    const float* lin_b      = (const float*)d_in[27];
    const float* bn3_g      = (const float*)d_in[28];
    const float* bn3_b      = (const float*)d_in[29];

    char* ws = (char*)d_ws;
    size_t off = 0;
    auto alloc = [&](size_t bytes) { size_t o = off; off += (bytes + 255) & ~(size_t)255; return o; };

    size_t o_xb    = alloc((size_t)BNr * 320 * 2);
    size_t o_eab   = alloc((size_t)BEr * 64 * 2);
    size_t o_w13   = alloc(256 * 320 * 2);
    size_t o_w23   = alloc(256 * 64 * 2);
    size_t o_inp   = alloc(768 * 256 * 2);
    size_t o_outp  = alloc(256 * 256 * 2);
    size_t o_l1    = alloc(1024 * 256 * 2);
    size_t o_l2    = alloc(256 * 1024 * 2);
    size_t o_sb    = alloc(256 * 4);
    size_t o_er    = alloc((size_t)BEr * 4);
    size_t o_ec    = alloc((size_t)BEr * 4);
    size_t o_deg   = alloc((size_t)BNr * 4);
    size_t o_dinv  = alloc((size_t)BNr * 4);
    size_t o_stats = alloc(2048 * 4);
    size_t o_Y1    = alloc((size_t)BNr * 256 * 4);
    size_t o_agg   = alloc((size_t)BNr * 256 * 4);
    size_t o_h     = alloc((size_t)BNr * 256 * 4);
    size_t o_hbnf  = alloc((size_t)BNr * 256 * 4);
    size_t o_hbnh  = alloc((size_t)BNr * 256 * 2);
    size_t o_q     = alloc((size_t)Bc * NHc * 512 * 64 * 2);
    size_t o_k     = alloc((size_t)Bc * NHc * 512 * 64 * 2);
    size_t o_vt    = alloc((size_t)Bc * NHc * 512 * 64 * 2);
    size_t o_attnb = alloc((size_t)BNr * 256 * 2);
    size_t o_h1h   = alloc((size_t)BNr * 256 * 2);
    size_t o_ffh   = alloc((size_t)BNr * 1024 * 2);
    size_t o_pool  = alloc(64 * 512 * 4);

    unsigned short* xb    = (unsigned short*)(ws + o_xb);
    unsigned short* eab   = (unsigned short*)(ws + o_eab);
    unsigned short* w13   = (unsigned short*)(ws + o_w13);
    unsigned short* w23   = (unsigned short*)(ws + o_w23);
    unsigned short* inph  = (unsigned short*)(ws + o_inp);
    unsigned short* outph = (unsigned short*)(ws + o_outp);
    unsigned short* l1h   = (unsigned short*)(ws + o_l1);
    unsigned short* l2h   = (unsigned short*)(ws + o_l2);
    float* sb    = (float*)(ws + o_sb);
    int*   er    = (int*)(ws + o_er);
    int*   ec    = (int*)(ws + o_ec);
    float* deg   = (float*)(ws + o_deg);
    float* dinv  = (float*)(ws + o_dinv);
    float* stats = (float*)(ws + o_stats);
    float* sums1 = stats, *sumsq1 = stats + 256, *scale1 = stats + 512, *shift1 = stats + 768;
    float* sums2 = stats + 1024, *sumsq2 = stats + 1280, *scale2 = stats + 1536, *shift2 = stats + 1792;
    float* Y1    = (float*)(ws + o_Y1);
    float* agg   = (float*)(ws + o_agg);
    float* h     = (float*)(ws + o_h);
    float* hbnf  = (float*)(ws + o_hbnf);
    unsigned short* hbnh  = (unsigned short*)(ws + o_hbnh);
    unsigned short* qb    = (unsigned short*)(ws + o_q);
    unsigned short* kb    = (unsigned short*)(ws + o_k);
    unsigned short* vtb   = (unsigned short*)(ws + o_vt);
    unsigned short* attnb = (unsigned short*)(ws + o_attnb);
    unsigned short* h1h   = (unsigned short*)(ws + o_h1h);
    unsigned short* ffh   = (unsigned short*)(ws + o_ffh);
    float* pool = (float*)(ws + o_pool);
    // reuse dead fp32 buffers
    float* tmp  = Y1;    // out_proj(+resid) output -> LN1 input (Y1 dead by then)
    float* h1f  = agg;   // LN1 output (agg dead)
    float* tmp2 = h;     // lin2(+resid) output (h dead)
    float* h2f  = hbnf;  // LN2 output (hbnf dead)

    dim3 blk(256);

    // ---- init ----
    k_zero<<<CEILDIV(BNr * 256, 256), blk, 0, stream>>>(agg, BNr * 256);
    k_zero<<<8, blk, 0, stream>>>(stats, 2048);
    k_fill<<<CEILDIV(BNr, 256), blk, 0, stream>>>(deg, 1.0f, BNr);

    // ---- precompute folded weights / casts ----
    k_pad_x<<<CEILDIV(BNr * 320, 256), blk, 0, stream>>>(x, xb);
    k_cast<<<CEILDIV(BEr * 64, 256), blk, 0, stream>>>(edge_attr, eab, BEr * 64);
    k_combine13<<<CEILDIV(256 * 320, 256), blk, 0, stream>>>(W3, W1, w13);
    k_combine23<<<CEILDIV(256 * 64, 256), blk, 0, stream>>>(W3, W2, w23);
    k_selfbias<<<1, blk, 0, stream>>>(W3, b3, sb);
    k_cast<<<CEILDIV(768 * 256, 256), blk, 0, stream>>>(in_proj_w, inph, 768 * 256);
    k_cast<<<CEILDIV(256 * 256, 256), blk, 0, stream>>>(out_proj_w, outph, 256 * 256);
    k_cast<<<CEILDIV(1024 * 256, 256), blk, 0, stream>>>(lin1_w, l1h, 1024 * 256);
    k_cast<<<CEILDIV(256 * 1024, 256), blk, 0, stream>>>(lin2_w, l2h, 256 * 1024);
    k_edge_prep<<<CEILDIV(BEr, 256), blk, 0, stream>>>(edge_row, edge_col, er, ec, deg);
    k_dinv<<<CEILDIV(BNr, 256), blk, 0, stream>>>(deg, dinv, BNr);

    // ---- GCN: Y1 = x @ (W3a.W1)^T ----
    gemm_bf16<0><<<dim3(2, BNr / 64), blk, 0, stream>>>(
        xb, 320, w13, 320, nullptr, Y1, nullptr, 256,
        nullptr, nullptr, nullptr, nullptr, nullptr, nullptr, nullptr,
        nullptr, nullptr, nullptr, 320);
    // edge GEMM + fused scatter: agg[col] += norm*(ea@(W3b.W2)^T + Y1[row] + b3)
    gemm_bf16<3><<<dim3(2, BEr / 64), blk, 0, stream>>>(
        eab, 64, w23, 64, nullptr, nullptr, nullptr, 256,
        nullptr, er, ec, dinv, agg, Y1, b3,
        nullptr, nullptr, nullptr, 64);
    k_gcn_final<<<CEILDIV(BNr * 256, 256), blk, 0, stream>>>(agg, Y1, sb, deg, dinv, gcn_bias, lengths, h);

    // ---- BN1 ----
    k_bn_stats<<<BNr / 128, blk, 0, stream>>>(h, sums1, sumsq1);
    k_bn_fin<<<1, blk, 0, stream>>>(sums1, sumsq1, bn1_g, bn1_b, scale1, shift1, 1.f / (float)BNr);
    k_bn_apply<<<CEILDIV(BNr * 256, 256), blk, 0, stream>>>(h, scale1, shift1, hbnf, hbnh);

    // ---- attention ----
    gemm_bf16<4><<<dim3(6, BNr / 64), blk, 0, stream>>>(
        hbnh, 256, inph, 256, in_proj_b, nullptr, nullptr, 0,
        nullptr, nullptr, nullptr, nullptr, nullptr, nullptr, nullptr,
        qb, kb, vtb, 256);
    k_attention<<<Bc * NHc * 32, dim3(32), 0, stream>>>(qb, kb, vtb, attnb);
    gemm_bf16<1><<<dim3(2, BNr / 64), blk, 0, stream>>>(
        attnb, 256, outph, 256, out_proj_b, tmp, nullptr, 256,
        hbnf, nullptr, nullptr, nullptr, nullptr, nullptr, nullptr,
        nullptr, nullptr, nullptr, 256);
    k_layernorm<<<BNr, blk, 0, stream>>>(tmp, ln1_g, ln1_b, h1f, h1h);

    // ---- FFN ----
    gemm_bf16<2><<<dim3(8, BNr / 64), blk, 0, stream>>>(
        h1h, 256, l1h, 256, lin1_b, nullptr, ffh, 1024,
        nullptr, nullptr, nullptr, nullptr, nullptr, nullptr, nullptr,
        nullptr, nullptr, nullptr, 256);
    gemm_bf16<1><<<dim3(2, BNr / 64), blk, 0, stream>>>(
        ffh, 1024, l2h, 1024, lin2_b, tmp2, nullptr, 256,
        h1f, nullptr, nullptr, nullptr, nullptr, nullptr, nullptr,
        nullptr, nullptr, nullptr, 1024);
    k_layernorm<<<BNr, blk, 0, stream>>>(tmp2, ln2_g, ln2_b, h2f, nullptr);

    // ---- BN2 + pooling + head ----
    k_bn_stats<<<BNr / 128, blk, 0, stream>>>(h2f, sums2, sumsq2);
    k_bn_fin<<<1, blk, 0, stream>>>(sums2, sumsq2, bn2_g, bn2_b, scale2, shift2, 1.f / (float)BNr);
    k_pool<<<64, blk, 0, stream>>>(h2f, scale2, shift2, lengths, pool);
    k_final<<<1, dim3(128), 0, stream>>>(pool, lin_w, lin_b, bn3_g, bn3_b, (float*)d_out);

    (void)in_sizes; (void)n_in; (void)out_size; (void)ws_size;
}
